// Generator_40621800685879
// MI455X (gfx1250) — matrix operand order repaired and checked by reading, not compiled, not run on hardware
//
#include <hip/hip_runtime.h>
#include <math.h>

typedef __attribute__((ext_vector_type(16))) _Float16 v16h;
typedef __attribute__((ext_vector_type(8)))  float    v8f;

#define BATCH  16
#define SDIM   512
#define SQRT2F 1.41421356237309515f

// ---------------------------------------------------------------------------
// Mapping network
// ---------------------------------------------------------------------------
__global__ void pixelnorm_kernel(const float* __restrict__ z, float* __restrict__ out) {
    __shared__ float red[256];
    int b = blockIdx.x;
    int t = threadIdx.x;
    float ss = 0.f;
    for (int i = t; i < SDIM; i += 256) { float v = z[b * SDIM + i]; ss += v * v; }
    red[t] = ss; __syncthreads();
    for (int s = 128; s > 0; s >>= 1) { if (t < s) red[t] += red[t + s]; __syncthreads(); }
    float scale = rsqrtf(red[0] / (float)SDIM + 1e-8f);
    for (int i = t; i < SDIM; i += 256) out[b * SDIM + i] = z[b * SDIM + i] * scale;
}

__global__ void mapping_layer_kernel(const float* __restrict__ in,
                                     const float* __restrict__ W,
                                     const float* __restrict__ bias,
                                     float* __restrict__ out) {
    int idx = blockIdx.x * blockDim.x + threadIdx.x;   // BATCH*SDIM
    if (idx >= BATCH * SDIM) return;
    int b = idx >> 9, o = idx & 511;
    const float scale = 0.01f / 22.62741699796952f;    // lr_mul / sqrt(512)
    const float* xr = in + b * SDIM;
    const float* wr = W + o * SDIM;
    float acc = 0.f;
    for (int i = 0; i < SDIM; ++i) acc += xr[i] * wr[i];
    float v = acc * scale + bias[o] * 0.01f;
    v = (v > 0.f) ? v : 0.2f * v;                      // fused leaky relu
    out[idx] = v * SQRT2F;
}

// s[b,ci] = (w[b,:] . mod_w[ci,:]) / sqrt(512) + mod_b[ci]
__global__ void style_kernel(const float* __restrict__ w,
                             const float* __restrict__ modw,
                             const float* __restrict__ modb,
                             float* __restrict__ s, int Cin) {
    int idx = blockIdx.x * blockDim.x + threadIdx.x;
    if (idx >= BATCH * Cin) return;
    int b = idx / Cin, c = idx - b * Cin;
    const float* wr = w + b * SDIM;
    const float* mr = modw + c * SDIM;
    float acc = 0.f;
    for (int i = 0; i < SDIM; ++i) acc += wr[i] * mr[i];
    s[idx] = acc * (1.0f / 22.62741699796952f) + modb[c];
}

// broadcast fp32 constant input into f16 feature buffer
__global__ void replicate_h_kernel(const float* __restrict__ src,
                                   _Float16* __restrict__ dst, int n) {
    int idx = blockIdx.x * blockDim.x + threadIdx.x;
    if (idx >= BATCH * n) return;
    dst[idx] = (_Float16)src[idx % n];
}

// ---------------------------------------------------------------------------
// Modulated 3x3 conv as implicit GEMM with WMMA f16 -> f32, tap-decomposed:
//   9 unrolled taps, K = Cin inside each tap (no k/9 decode, immediate strides).
//   NT=4 spatial sub-tiles per wave: the A fragment (modulated weights, the
//   expensive gather) is loaded once per K-step and reused by 4 WMMAs.
//   MODE 0 : same-pad conv (pad 1), OH=H, OW=W
//   MODE 1 : conv_transpose stride 2, OH=2H+1, OW=2W+1
// Fragment layouts per CDNA5 ISA 7.12.2:
//   A (16x32 f16): lane = m + 16*((k%16)/8), half j = (k/16)*8 + (k%8)
//   B (32x16 f16): lane = n + 16*((k%16)/8), half j = (k/16)*8 + (k%8)
//   C (16x16 f32): lane = n + 16*(m/8),      vgpr i = m%8
// ---------------------------------------------------------------------------
template<int H, int W, int OH, int OW, int MODE>
__global__ void __launch_bounds__(32)
conv3x3_wmma_kernel(const _Float16* __restrict__ in, const float* __restrict__ weight,
                    const float* __restrict__ s, float* __restrict__ out,
                    int Cin, int Cout, float wscale) {
    constexpr int NT   = 4;
    constexpr int HWc  = H * W;
    constexpr int OHWc = OH * OW;
    __shared__ float s_sm[512];
    const int lane = threadIdx.x;
    const int b = blockIdx.z;
    for (int i = lane; i < Cin; i += 32) s_sm[i] = wscale * s[b * Cin + i];
    __syncthreads();

    const int m0 = blockIdx.x * 16;
    const int n0 = blockIdx.y * (16 * NT);

    const int mrow  = lane & 15;          // A: Cout row this lane holds
    const int kmid  = lane >> 4;          // A: which 8-wide k sub-band
    const int khalf = lane >> 4;          // B: which 8-wide k sub-band

    int  oyv[NT], oxv[NT];
    bool nv[NT];
#pragma unroll
    for (int u = 0; u < NT; ++u) {
        int ncol = n0 + u * 16 + (lane & 15);
        nv[u]  = (ncol < OHWc);
        oyv[u] = nv[u] ? (ncol / OW) : 0;
        oxv[u] = nv[u] ? (ncol - oyv[u] * OW) : 0;
    }

    const float*    wrow = weight + (size_t)(m0 + mrow) * Cin * 9;
    const _Float16* inb  = in + (size_t)b * Cin * HWc;

    v8f c[NT];
#pragma unroll
    for (int u = 0; u < NT; ++u) c[u] = (v8f){};

#pragma unroll
    for (int t = 0; t < 9; ++t) {
        const int ky = t / 3, kx = t - ky * 3;
        // ---- per-tap spatial validity + base offsets (loop invariant in ci) ----
        bool bval[NT];
        const _Float16* bq[NT];
#pragma unroll
        for (int u = 0; u < NT; ++u) {
            bool v = nv[u];
            int base_off = 0;
            if (MODE == 0) {
                int iy = oyv[u] + ky - 1, ix = oxv[u] + kx - 1;
                v = v && (iy >= 0) && (iy < H) && (ix >= 0) && (ix < W);
                if (v) base_off = iy * W + ix;
            } else {
                int ty = oyv[u] - ky, tx = oxv[u] - kx;
                v = v && (ty >= 0) && (tx >= 0) && !(ty & 1) && !(tx & 1);
                int iy = ty >> 1, ix = tx >> 1;
                v = v && (iy < H) && (ix < W);
                if (v) base_off = iy * W + ix;
            }
            bval[u] = v;
            bq[u] = inb + base_off + (size_t)(khalf * 8) * HWc;
        }
        const float* wp = wrow + t + kmid * 72;      // kmid*8 rows * 9 taps
        const float* sp = s_sm + kmid * 8;

        for (int ci0 = 0; ci0 < Cin; ci0 += 32) {
            const float* wq = wp + (size_t)ci0 * 9;
            __builtin_prefetch(wq + 32 * 9, 0, 1);   // next K-step weight rows
            v16h a;
#pragma unroll
            for (int j = 0; j < 16; ++j) {
                const int e = ((j >> 3) << 4) + (j & 7);  // 0..7, 16..23
                a[j] = (_Float16)(wq[e * 9] * sp[ci0 + e]);
            }
#pragma unroll
            for (int u = 0; u < NT; ++u) {
                const _Float16* bq2 = bq[u] + (size_t)ci0 * HWc;
                v16h bb;
#pragma unroll
                for (int j = 0; j < 16; ++j) {
                    _Float16 bv = bq2[(size_t)(((j >> 3) << 4) + (j & 7)) * HWc];
                    bb[j] = bval[u] ? bv : (_Float16)0.f;
                }
                c[u] = __builtin_amdgcn_wmma_f32_16x16x32_f16(false, a, false, bb,
                                                              (short)0, c[u], false, false);
            }
        }
    }

    // ---- store C tiles ----
    const int mb = (lane >> 4) * 8;
#pragma unroll
    for (int u = 0; u < NT; ++u) {
        if (nv[u]) {
#pragma unroll
            for (int i = 0; i < 8; ++i) {
                int co = m0 + mb + i;
                out[(((size_t)b * Cout + co) * OH + oyv[u]) * OW + oxv[u]] = c[u][i];
            }
        }
    }
}

// per-(b,co): divide by unbiased std over spatial, add act bias, lrelu*sqrt2;
// reads f32 conv output, writes f16 activation buffer
__global__ void demod_lrelu_kernel(const float* __restrict__ x,
                                   _Float16* __restrict__ oh,
                                   const float* __restrict__ actb,
                                   int Cout, int HW) {
    __shared__ float rsum[256], rsq[256];
    int bc = blockIdx.x;                 // b*Cout + co
    int co = bc % Cout;
    const float* p = x + (size_t)bc * HW;
    _Float16* q = oh + (size_t)bc * HW;
    int t = threadIdx.x;
    float s0 = 0.f, s1 = 0.f;
    for (int i = t; i < HW; i += 256) { float v = p[i]; s0 += v; s1 += v * v; }
    rsum[t] = s0; rsq[t] = s1; __syncthreads();
    for (int s = 128; s > 0; s >>= 1) {
        if (t < s) { rsum[t] += rsum[t + s]; rsq[t] += rsq[t + s]; }
        __syncthreads();
    }
    float mean = rsum[0] / (float)HW;
    float var  = (rsq[0] - mean * rsum[0]) / (float)(HW - 1);
    float inv  = rsqrtf(fmaxf(var, 1e-20f));
    float ab   = actb[co];
    for (int i = t; i < HW; i += 256) {
        float v = p[i] * inv + ab;
        v = (v > 0.f) ? v : 0.2f * v;
        q[i] = (_Float16)(v * SQRT2F);
    }
}

// upfirdn2d with 4x4 blur*4 kernel: out[i,j] = sum kv[u]kv[v]/16 * Dil[i+u-pad, j+v-pad]
__global__ void upfirdn_kernel(const float* __restrict__ in, float* __restrict__ out,
                               int C, int H, int W, int OH, int OW, int up, int pad0) {
    int idx = blockIdx.x * blockDim.x + threadIdx.x;
    int total = BATCH * C * OH * OW;
    if (idx >= total) return;
    int ox = idx % OW; int r = idx / OW;
    int oyv = r % OH;  int bc = r / OH;
    const float kv[4] = {1.f, 3.f, 3.f, 1.f};
    const float* src = in + (size_t)bc * H * W;
    float acc = 0.f;
#pragma unroll
    for (int u = 0; u < 4; ++u) {
        int py = oyv + u - pad0;
        if (py < 0 || (py % up)) continue;
        int iy = py / up; if (iy >= H) continue;
#pragma unroll
        for (int v = 0; v < 4; ++v) {
            int px = ox + v - pad0;
            if (px < 0 || (px % up)) continue;
            int ix = px / up; if (ix >= W) continue;
            acc += kv[u] * kv[v] * src[(size_t)iy * W + ix];
        }
    }
    out[idx] = acc * (1.0f / 16.0f);
}

// to_xyz 1x1 modulated conv (no demod); optionally accumulates into out (skip add)
__global__ void toxyz_kernel(const _Float16* __restrict__ x, const float* __restrict__ w1,
                             const float* __restrict__ s, const float* __restrict__ outb,
                             float* __restrict__ out, int Cin, int HW, float scale,
                             int addflag) {
    int idx = blockIdx.x * blockDim.x + threadIdx.x;
    int total = BATCH * 3 * HW;
    if (idx >= total) return;
    int p = idx % HW; int r = idx / HW;
    int o = r % 3;    int b = r / 3;
    const _Float16* xb = x + (size_t)b * Cin * HW + p;
    const float* wr = w1 + o * Cin;
    const float* sb = s + b * Cin;
    float acc = 0.f;
    for (int ci = 0; ci < Cin; ++ci) acc += wr[ci] * sb[ci] * (float)xb[(size_t)ci * HW];
    float v = acc * scale + outb[o];
    out[idx] = addflag ? (out[idx] + v) : v;
}

// ---------------------------------------------------------------------------
template<int H, int W, int OH, int OW, int MODE>
static inline void launch_conv(const _Float16* in, const float* w, const float* s,
                               float* out, int Cin, int Cout, float wscale,
                               hipStream_t stream) {
    conv3x3_wmma_kernel<H, W, OH, OW, MODE>
        <<<dim3(Cout / 16, (OH * OW + 63) / 64, BATCH), 32, 0, stream>>>(
            in, w, s, out, Cin, Cout, wscale);
}

extern "C" void kernel_launch(void* const* d_in, const int* in_sizes, int n_in,
                              void* d_out, int out_size, void* d_ws, size_t ws_size,
                              hipStream_t stream) {
    (void)in_sizes; (void)n_in; (void)out_size; (void)ws_size;
    const float* z   = (const float*)d_in[0];
    const float* cst = (const float*)d_in[17];
    const int CH[5] = {512, 512, 512, 256, 128};

    // ---- workspace layout (bytes) ----
    char* base = (char*)d_ws;
    size_t off = 0;
    float* wA    = (float*)(base + off); off += (size_t)BATCH * SDIM * 4;
    float* wB    = (float*)(base + off); off += (size_t)BATCH * SDIM * 4;
    float* sb    = (float*)(base + off); off += (size_t)BATCH * SDIM * 4;
    float* skipA = (float*)(base + off); off += (size_t)BATCH * 3 * 64 * 64 * 4;
    float* skipB = (float*)(base + off); off += (size_t)BATCH * 3 * 64 * 64 * 4;
    float* xF    = (float*)(base + off); off += (size_t)BATCH * 524288 * 4;      // f32 post-blur
    float* tmp   = (float*)(base + off); off += (size_t)BATCH * 128 * 65 * 65 * 4; // f32 conv out
    _Float16* xH = (_Float16*)(base + off); off += (size_t)BATCH * 524288 * 2;   // f16 features
    _Float16* yH = (_Float16*)(base + off); off += (size_t)BATCH * 524288 * 2;

    // ---- mapping network ----
    pixelnorm_kernel<<<BATCH, 256, 0, stream>>>(z, wA);
    float* win = wA; float* wout = wB;
    for (int l = 0; l < 8; ++l) {
        mapping_layer_kernel<<<(BATCH * SDIM + 255) / 256, 256, 0, stream>>>(
            win, (const float*)d_in[1 + 2 * l], (const float*)d_in[2 + 2 * l], wout);
        float* t = win; win = wout; wout = t;
    }
    const float* wlat = win;

    // ---- constant input (f16) ----
    replicate_h_kernel<<<(BATCH * 8192 + 255) / 256, 256, 0, stream>>>(cst, xH, 8192);

    // ---- conv1 (4x4, 512->512, same) ----
    style_kernel<<<(BATCH * 512 + 255) / 256, 256, 0, stream>>>(
        wlat, (const float*)d_in[19], (const float*)d_in[20], sb, 512);
    launch_conv<4, 4, 4, 4, 0>(xH, (const float*)d_in[18], sb, tmp, 512, 512,
                               1.0f / sqrtf(512.f * 9.f), stream);
    demod_lrelu_kernel<<<BATCH * 512, 256, 0, stream>>>(
        tmp, xH, (const float*)d_in[21], 512, 16);   // const input dead -> reuse xH

    // ---- to_xyz[0] ----
    style_kernel<<<(BATCH * 512 + 255) / 256, 256, 0, stream>>>(
        wlat, (const float*)d_in[55], (const float*)d_in[56], sb, 512);
    toxyz_kernel<<<(BATCH * 3 * 16 + 255) / 256, 256, 0, stream>>>(
        xH, (const float*)d_in[54], sb, (const float*)d_in[57],
        skipA, 512, 16, 1.0f / sqrtf(512.f), 0);

    float* skip_cur = skipA;
    for (int st = 0; st < 4; ++st) {
        int cin = CH[st], cout = CH[st + 1];
        int Hc = 4 << st;          // input res
        int Hn = Hc * 2;           // output res
        int OHt = 2 * Hc + 1;      // tconv intermediate res
        const float* wup   = (const float*)d_in[22 + st * 8 + 0];
        const float* mwup  = (const float*)d_in[22 + st * 8 + 1];
        const float* mbup  = (const float*)d_in[22 + st * 8 + 2];
        const float* abup  = (const float*)d_in[22 + st * 8 + 3];
        const float* wkp   = (const float*)d_in[22 + st * 8 + 4];
        const float* mwkp  = (const float*)d_in[22 + st * 8 + 5];
        const float* mbkp  = (const float*)d_in[22 + st * 8 + 6];
        const float* abkp  = (const float*)d_in[22 + st * 8 + 7];
        const float* txw   = (const float*)d_in[54 + (st + 1) * 4 + 0];
        const float* txmw  = (const float*)d_in[54 + (st + 1) * 4 + 1];
        const float* txmb  = (const float*)d_in[54 + (st + 1) * 4 + 2];
        const float* txob  = (const float*)d_in[54 + (st + 1) * 4 + 3];
        float wsu = 1.0f / sqrtf((float)(cin * 9));
        float wsk = 1.0f / sqrtf((float)(cout * 9));

        // a) upsample tconv (xH -> tmp, f32)
        style_kernel<<<(BATCH * cin + 255) / 256, 256, 0, stream>>>(wlat, mwup, mbup, sb, cin);
        switch (st) {
        case 0: launch_conv<4, 4, 9, 9, 1>(xH, wup, sb, tmp, cin, cout, wsu, stream); break;
        case 1: launch_conv<8, 8, 17, 17, 1>(xH, wup, sb, tmp, cin, cout, wsu, stream); break;
        case 2: launch_conv<16, 16, 33, 33, 1>(xH, wup, sb, tmp, cin, cout, wsu, stream); break;
        default: launch_conv<32, 32, 65, 65, 1>(xH, wup, sb, tmp, cin, cout, wsu, stream); break;
        }
        // b) blur (tmp -> xF), size OHt -> Hn
        {
            int total = BATCH * cout * Hn * Hn;
            upfirdn_kernel<<<(total + 255) / 256, 256, 0, stream>>>(
                tmp, xF, cout, OHt, OHt, Hn, Hn, 1, 1);
        }
        // c) demod + lrelu: xF (f32) -> yH (f16)
        demod_lrelu_kernel<<<BATCH * cout, 256, 0, stream>>>(xF, yH, abup, cout, Hn * Hn);
        // d) keep conv (yH -> tmp, f32)
        style_kernel<<<(BATCH * cout + 255) / 256, 256, 0, stream>>>(wlat, mwkp, mbkp, sb, cout);
        switch (st) {
        case 0: launch_conv<8, 8, 8, 8, 0>(yH, wkp, sb, tmp, cout, cout, wsk, stream); break;
        case 1: launch_conv<16, 16, 16, 16, 0>(yH, wkp, sb, tmp, cout, cout, wsk, stream); break;
        case 2: launch_conv<32, 32, 32, 32, 0>(yH, wkp, sb, tmp, cout, cout, wsk, stream); break;
        default: launch_conv<64, 64, 64, 64, 0>(yH, wkp, sb, tmp, cout, cout, wsk, stream); break;
        }
        // e) demod + lrelu: tmp (f32) -> xH (f16), old features dead
        demod_lrelu_kernel<<<BATCH * cout, 256, 0, stream>>>(tmp, xH, abkp, cout, Hn * Hn);
        // f) skip upsample (up=2, pad 2) into next skip buffer (or d_out on last stage)
        float* skip_next = (st == 3) ? (float*)d_out
                                     : ((skip_cur == skipA) ? skipB : skipA);
        {
            int total = BATCH * 3 * Hn * Hn;
            upfirdn_kernel<<<(total + 255) / 256, 256, 0, stream>>>(
                skip_cur, skip_next, 3, Hc, Hc, Hn, Hn, 2, 2);
        }
        // g) to_xyz on new features, accumulated into upsampled skip
        style_kernel<<<(BATCH * cout + 255) / 256, 256, 0, stream>>>(wlat, txmw, txmb, sb, cout);
        toxyz_kernel<<<(BATCH * 3 * Hn * Hn + 255) / 256, 256, 0, stream>>>(
            xH, txw, sb, txob, skip_next, cout, Hn * Hn,
            1.0f / sqrtf((float)cout), 1);
        skip_cur = skip_next;
    }
}
